// MoChA_28939489640849
// MI455X (gfx1250) — matrix operand order repaired and checked by reading, not compiled, not run on hardware
//
#include <hip/hip_runtime.h>
#include <stdint.h>

#define NEG_INF (-3.4028234663852886e38f)

typedef __attribute__((ext_vector_type(16))) __bf16 v16bf;
typedef __attribute__((ext_vector_type(8)))  float  v8f;

union V16U { v16bf v; uint4 q[2]; };

__device__ __forceinline__ unsigned short f2bf(float f) {
    unsigned int u = __float_as_uint(f);
    unsigned int r = u + 0x7FFFu + ((u >> 16) & 1u);   // round-to-nearest-even
    return (unsigned short)(r >> 16);
}

// ---------------------------------------------------------------------------
// Kernel: query projections  qproj[b][n], n in [0,1024): 0-511 mono, 512-1023 chunk
// ---------------------------------------------------------------------------
__global__ __launch_bounds__(256) void mocha_qproj(
    const float* __restrict__ query, const float* __restrict__ wqm,
    const float* __restrict__ wqc, float* __restrict__ qproj) {
    int b = blockIdx.x;
    int n = blockIdx.y * 256 + threadIdx.x;   // 0..1023
    const float* w = (n < 512) ? wqm : wqc;
    int a = n & 511;
    const float* q = query + b * 512;
    float s = 0.f;
    for (int k = 0; k < 512; ++k) s += q[k] * w[k * 512 + a];
    qproj[b * 1024 + n] = s;
}

// ---------------------------------------------------------------------------
// Kernel: s = g / ||v_mono||
// ---------------------------------------------------------------------------
__global__ __launch_bounds__(256) void mocha_vnorm(
    const float* __restrict__ vm, const float* __restrict__ g,
    float* __restrict__ s_out) {
    __shared__ float red[256];
    int tid = threadIdx.x;
    float x = vm[tid], y = vm[tid + 256];
    red[tid] = x * x + y * y;
    __syncthreads();
    for (int off = 128; off; off >>= 1) {
        if (tid < off) red[tid] += red[tid + off];
        __syncthreads();
    }
    if (tid == 0) s_out[0] = g[0] / sqrtf(red[0]);
}

// ---------------------------------------------------------------------------
// Kernel: fused energy GEMM via WMMA bf16.
// grid = (125 row-groups, 8 column-slices of 128), block = 256 (8 waves).
// Weight slice resident in LDS (bf16); B fragments hoisted into registers
// (strip-invariant); next strip's key rows register-prefetched so HBM latency
// overlaps the 16-step WMMA chain. Deterministic (no atomics).
// e_part layout: [slice][64000] ; slices 0-3 sum -> e_mono, 4-7 -> e_chunk.
// ---------------------------------------------------------------------------
__global__ __launch_bounds__(256) void mocha_energy(
    const float* __restrict__ key,
    const float* __restrict__ wkm, const float* __restrict__ bkm,
    const float* __restrict__ vm,
    const float* __restrict__ wkc, const float* __restrict__ bkc,
    const float* __restrict__ vc,
    const float* __restrict__ qproj,
    float* __restrict__ e_part) {
    extern __shared__ char smem[];
    unsigned short* wl = (unsigned short*)smem;      // 128 x 520 bf16 (padded rows)
    unsigned short* al = wl + 128 * 520;             // 16 x 520 bf16
    float* estrip = (float*)(al + 16 * 520);         // 8 waves x 16 rows

    const int tid   = threadIdx.x;
    const int slice = blockIdx.y;                    // 0..7
    const int bx    = blockIdx.x;                    // 0..124
    const int lane  = tid & 31;
    const int wave  = tid >> 5;
    const int hv    = lane >> 4;                     // half-wave 0/1
    const int l15   = lane & 15;

    // ---- stage weight slice (f32 -> bf16) ----
    const float* wsrc = (slice < 4) ? wkm : wkc;
    const int colbase = (slice & 3) * 128;
    for (int i = 0; i < 256; ++i) {
        int idx = tid + i * 256;                     // 65536 elements
        int nl  = idx & 127;
        int k   = idx >> 7;
        wl[nl * 520 + k] = f2bf(wsrc[k * 512 + colbase + nl]);
    }

    // per-lane column constants for this wave's output tile
    const int nl_lane = wave * 16 + l15;             // 0..127
    const int ncol    = slice * 128 + nl_lane;       // 0..1023 (concat space)
    const float bias  = ((slice < 4) ? bkm : bkc)[colbase + nl_lane];
    const float vv    = ((slice < 4) ? vm  : vc )[colbase + nl_lane];
    __syncthreads();

    // ---- hoist B fragments into registers (invariant across strips) ----
    const unsigned short* brow = wl + nl_lane * 520;
    V16U bfrag[16];
#pragma unroll
    for (int kc = 0; kc < 16; ++kc) {
        int kb = kc * 32 + hv * 16;                  // B: contiguous 16 K per half
        bfrag[kc].q[0] = *(const uint4*)(brow + kb);
        bfrag[kc].q[1] = *(const uint4*)(brow + kb + 8);
    }

    const unsigned short* arow = al + l15 * 520;

    // ---- prologue: prefetch strip 0's key rows into registers ----
    float kreg[32];
    {
        const float* k0 = key + (size_t)(bx * 32) * 16 * 512;
#pragma unroll
        for (int i = 0; i < 32; ++i) kreg[i] = k0[tid + i * 256];
    }

    for (int s = 0; s < 32; ++s) {
        int g = bx * 32 + s;                         // strip 0..3999
        int b = g / 250;
        size_t rowbase = (size_t)g * 16;

        // ---- convert prefetched rows: relu -> bf16 -> LDS ----
#pragma unroll
        for (int i = 0; i < 32; ++i) {
            int idx = tid + i * 256;                 // 8192 elements
            al[(idx >> 9) * 520 + (idx & 511)] = f2bf(fmaxf(kreg[i], 0.0f));
        }
        __syncthreads();

        // ---- prefetch next strip (clamped at the end) while WMMAs run ----
        {
            int gn = (g < 3999) ? g + 1 : g;
            const float* kn = key + (size_t)gn * 16 * 512;
#pragma unroll
            for (int i = 0; i < 32; ++i) kreg[i] = kn[tid + i * 256];
        }

        // ---- K loop: 16 x wmma_f32_16x16x32_bf16 (A from LDS, B in regs) ----
        v8f acc = {0.f, 0.f, 0.f, 0.f, 0.f, 0.f, 0.f, 0.f};
#pragma unroll
        for (int kc = 0; kc < 16; ++kc) {
            V16U a;
            int ka = kc * 32 + hv * 16;              // A: contiguous 16 K per half
            a.q[0] = *(const uint4*)(arow + ka);
            a.q[1] = *(const uint4*)(arow + ka + 8);
            acc = __builtin_amdgcn_wmma_f32_16x16x32_bf16(
                false, a.v, false, bfrag[kc].v, (short)0, acc, false, false);
        }

        // ---- tanh + v-dot, reduce across 16 lanes per half ----
        float qp = qproj[b * 1024 + ncol];
        float red[8];
#pragma unroll
        for (int r = 0; r < 8; ++r)
            red[r] = vv * tanhf(acc[r] + bias + qp);
#pragma unroll
        for (int m = 1; m < 16; m <<= 1) {
#pragma unroll
            for (int r = 0; r < 8; ++r)
                red[r] += __shfl_xor(red[r], m, 32);
        }
        if (l15 == 0) {
#pragma unroll
            for (int r = 0; r < 8; ++r)
                estrip[wave * 16 + hv * 8 + r] = red[r];
        }
        __syncthreads();
        if (tid < 16) {                              // fixed-order sum -> deterministic
            float sum = 0.f;
            for (int w = 0; w < 8; ++w) sum += estrip[w * 16 + tid];
            e_part[(size_t)slice * 64000 + rowbase + tid] = sum;
        }
        __syncthreads();
    }
}

// ---------------------------------------------------------------------------
// Threefry-2x32 (JAX layout) + erfinv for jax.random.normal(key(123))
// ---------------------------------------------------------------------------
__device__ __forceinline__ unsigned rotl32(unsigned v, int d) {
    return (v << d) | (v >> (32 - d));
}
__device__ __forceinline__ void threefry2x32(unsigned k0, unsigned k1,
                                             unsigned x0, unsigned x1,
                                             unsigned& o0, unsigned& o1) {
    unsigned ks2 = k0 ^ k1 ^ 0x1BD11BDAu;
    x0 += k0; x1 += k1;
#define TF_RND(r) { x0 += x1; x1 = rotl32(x1, r); x1 ^= x0; }
    TF_RND(13) TF_RND(15) TF_RND(26) TF_RND(6)
    x0 += k1;  x1 += ks2 + 1u;
    TF_RND(17) TF_RND(29) TF_RND(16) TF_RND(24)
    x0 += ks2; x1 += k0 + 2u;
    TF_RND(13) TF_RND(15) TF_RND(26) TF_RND(6)
    x0 += k0;  x1 += k1 + 3u;
    TF_RND(17) TF_RND(29) TF_RND(16) TF_RND(24)
    x0 += k1;  x1 += ks2 + 4u;
    TF_RND(13) TF_RND(15) TF_RND(26) TF_RND(6)
    x0 += ks2; x1 += k0 + 5u;
#undef TF_RND
    o0 = x0; o1 = x1;
}
__device__ __forceinline__ float erfinv_f(float x) {
    float w = -logf((1.0f - x) * (1.0f + x));
    float p;
    if (w < 5.0f) {
        w -= 2.5f;
        p = 2.81022636e-08f;
        p = fmaf(p, w, 3.43273939e-07f);
        p = fmaf(p, w, -3.5233877e-06f);
        p = fmaf(p, w, -4.39150654e-06f);
        p = fmaf(p, w, 0.00021858087f);
        p = fmaf(p, w, -0.00125372503f);
        p = fmaf(p, w, -0.00417768164f);
        p = fmaf(p, w, 0.246640727f);
        p = fmaf(p, w, 1.50140941f);
    } else {
        w = sqrtf(w) - 3.0f;
        p = -0.000200214257f;
        p = fmaf(p, w, 0.000100950558f);
        p = fmaf(p, w, 0.00134934322f);
        p = fmaf(p, w, -0.00367342844f);
        p = fmaf(p, w, 0.00573950773f);
        p = fmaf(p, w, -0.0076224613f);
        p = fmaf(p, w, 0.00943887047f);
        p = fmaf(p, w, 1.00167406f);
        p = fmaf(p, w, 2.83297682f);
    }
    return p * x;
}

// ---------------------------------------------------------------------------
// Kernel: per-batch scan. One block per batch b (256 threads, 250 active x 16).
// Computes p_choose, exclusive log-cumprod, alpha (-> d_out), chunk softmax,
// CHUNK=8 moving sums, beta (-> ws). Deterministic (fixed-order reductions).
// ---------------------------------------------------------------------------
__global__ __launch_bounds__(256) void mocha_scan(
    const float* __restrict__ e_part, const float* __restrict__ snorm,
    const float* __restrict__ rmono, const int* __restrict__ mask,
    float* __restrict__ out, float* __restrict__ beta_ws) {
    extern __shared__ char smem2[];
    float* alpha   = (float*)smem2;        // 4000
    float* se      = alpha + 4000;         // 4000
    float* arr2    = se + 4000;            // 4000
    float* partial = arr2 + 4000;          // 256

    const int b = blockIdx.x, tid = threadIdx.x;
    const float sc = snorm[0];
    const float r  = rmono[0];
    const size_t rb = (size_t)b * 4000;
    const bool act = (tid < 250);

    float pch[16], l1[16];
    float run = 0.f;
    if (act) {
        for (int j = 0; j < 16; ++j) {
            int t = tid * 16 + j;
            size_t row = rb + t;
            float em = e_part[row] + e_part[64000 + row] +
                       e_part[2 * 64000 + row] + e_part[3 * 64000 + row];
            em = sc * em + r;
            if (mask[row] == 0) em = NEG_INF;
            // jax.random.normal(key(123)) noise, std 1.0
            unsigned j32 = (unsigned)row;
            unsigned pi = j32 % 32000u, half_ = j32 / 32000u;
            unsigned o0, o1;
            threefry2x32(0u, 123u, pi, pi + 32000u, o0, o1);
            unsigned bits = half_ ? o1 : o0;
            float f = __uint_as_float(0x3f800000u | (bits >> 9)) - 1.0f;
            const float lo = -0.99999994f;
            float u = lo + f * (1.0f - lo);
            float noise = 1.41421356237f * erfinv_f(u);
            float p = 1.0f / (1.0f + expf(-(em + noise)));
            pch[j] = p;
            float c = fminf(fmaxf(1.0f - p, 1e-6f), 1.0f);
            float lg = logf(c);
            l1[j] = lg;
            run += lg;
        }
    }
    partial[tid] = run;
    __syncthreads();
    for (int off = 1; off < 256; off <<= 1) {   // inclusive scan of thread totals
        float add = (tid >= off) ? partial[tid - off] : 0.f;
        __syncthreads();
        partial[tid] += add;
        __syncthreads();
    }
    float pre = partial[tid] - run;
    if (act) {
        float runl = 0.f;
        for (int j = 0; j < 16; ++j) {
            int t = tid * 16 + j;
            float excl = pre + runl;            // exclusive cumsum of log1mp
            runl += l1[j];
            float a = pch[j] * expf(excl);      // alpha = p * cumprod_1mp
            alpha[t] = a;
            out[16 * 512 + rb + t] = a;         // alpha output
        }
    }

    // ---- chunk energies + max ----
    float lmax = NEG_INF;
    if (act) {
        for (int j = 0; j < 16; ++j) {
            int t = tid * 16 + j;
            size_t row = rb + t;
            float ec = e_part[4 * 64000 + row] + e_part[5 * 64000 + row] +
                       e_part[6 * 64000 + row] + e_part[7 * 64000 + row];
            if (mask[row] == 0) ec = NEG_INF;
            se[t] = ec;
            lmax = fmaxf(lmax, ec);
        }
    }
    __syncthreads();
    partial[tid] = lmax;
    __syncthreads();
    for (int off = 128; off; off >>= 1) {
        if (tid < off) partial[tid] = fmaxf(partial[tid], partial[tid + off]);
        __syncthreads();
    }
    float mx = partial[0];
    __syncthreads();
    if (act)
        for (int j = 0; j < 16; ++j) {
            int t = tid * 16 + j;
            se[t] = fmaxf(expf(se[t] - mx), 1e-5f);
        }
    __syncthreads();
    // denom over [t-7, t]; arr2 = alpha / denom
    if (act)
        for (int j = 0; j < 16; ++j) {
            int t = tid * 16 + j;
            float d = 0.f;
            int lo2 = (t - 7 > 0) ? t - 7 : 0;
            for (int tt = lo2; tt <= t; ++tt) d += se[tt];
            arr2[t] = alpha[t] / d;
        }
    __syncthreads();
    // beta = se * moving_sum(arr2, fwd 7)
    if (act)
        for (int j = 0; j < 16; ++j) {
            int t = tid * 16 + j;
            int hi2 = (t + 7 < 3999) ? t + 7 : 3999;
            float m2 = 0.f;
            for (int tt = t; tt <= hi2; ++tt) m2 += arr2[tt];
            beta_ws[rb + t] = se[t] * m2;
        }
}

// ---------------------------------------------------------------------------
// Kernel: cv[b][d] = sum_t beta[b,t] * value[b,t,d]  (deterministic, coalesced)
// ---------------------------------------------------------------------------
__global__ __launch_bounds__(512) void mocha_cv(
    const float* __restrict__ value, const float* __restrict__ beta,
    float* __restrict__ out) {
    int b = blockIdx.x;
    int d = threadIdx.x;                         // 512
    const float* vb = value + (size_t)b * 4000 * 512;
    const float* bb = beta + (size_t)b * 4000;
    float acc = 0.f;
    for (int t = 0; t < 4000; ++t)
        acc = fmaf(bb[t], vb[(size_t)t * 512 + d], acc);
    out[b * 512 + d] = acc;
}

// ---------------------------------------------------------------------------
extern "C" void kernel_launch(void* const* d_in, const int* in_sizes, int n_in,
                              void* d_out, int out_size, void* d_ws, size_t ws_size,
                              hipStream_t stream) {
    (void)in_sizes; (void)n_in; (void)out_size; (void)ws_size;
    const float* key   = (const float*)d_in[0];
    const float* value = (const float*)d_in[1];
    const float* query = (const float*)d_in[2];
    const int*   mask  = (const int*)d_in[3];
    const float* wkm   = (const float*)d_in[4];
    const float* bkm   = (const float*)d_in[5];
    const float* wqm   = (const float*)d_in[6];
    const float* vm    = (const float*)d_in[7];
    const float* gm    = (const float*)d_in[8];
    const float* rm    = (const float*)d_in[9];
    const float* wkc   = (const float*)d_in[10];
    const float* bkc   = (const float*)d_in[11];
    const float* wqc   = (const float*)d_in[12];
    const float* vc    = (const float*)d_in[13];
    float* out = (float*)d_out;
    float* ws  = (float*)d_ws;

    float* e_part = ws;                 // 8 * 64000
    float* qproj  = ws + 512000;        // 16 * 1024
    float* snorm  = ws + 528384;        // 1 (padded)
    float* beta   = ws + 528448;        // 64000

    mocha_qproj<<<dim3(16, 4), 256, 0, stream>>>(query, wqm, wqc, qproj);
    mocha_vnorm<<<1, 256, 0, stream>>>(vm, gm, snorm);

    size_t lds1 = (size_t)(128 * 520 + 16 * 520) * sizeof(unsigned short) +
                  8 * 16 * sizeof(float);
    mocha_energy<<<dim3(125, 8), 256, lds1, stream>>>(
        key, wkm, bkm, vm, wkc, bkc, vc, qproj, e_part);

    size_t lds2 = (size_t)(3 * 4000 + 256) * sizeof(float);
    mocha_scan<<<16, 256, lds2, stream>>>(e_part, snorm, rm, mask, out, beta);

    mocha_cv<<<16, 512, 0, stream>>>(value, beta, out);
}
